// VanillaRNN_13056700580615
// MI455X (gfx1250) — matrix operand order repaired and checked by reading, not compiled, not run on hardware
//
#include <hip/hip_runtime.h>
#include <hip/hip_bf16.h>

typedef __attribute__((ext_vector_type(16))) __bf16 v16bf;
typedef __attribute__((ext_vector_type(8)))  float  v8f;

#define T_STEPS 256
#define D_DIM   256
#define H_DIM   1024
#define C_DIM   1024
#define B_DIM   128

#define MT_H 64            // H/16 row tiles
#define KT_H 32            // H/32 k tiles
#define KT_D 8             // D/32 k tiles
#define NB   8             // B/16 batch column blocks
#define FRAG 512           // 32 lanes * 16 bf16 elems per fragment

// workspace offsets in unsigned short (bf16) elements
#define WS_WHX 0
#define WS_WHH (WS_WHX + MT_H*KT_D*FRAG)                 // Whx frags: 512KB
#define WS_WHY (WS_WHH + MT_H*KT_H*FRAG)                 // Whh frags: 2MB
#define WS_XF  (WS_WHY + MT_H*KT_H*FRAG)                 // Why frags: 2MB
#define WS_HF  (WS_XF  + (size_t)T_STEPS*KT_D*NB*FRAG)   // x frags: 16MB
// HF: 32*8*FRAG = 256KB; total ws ~21MB

__device__ __forceinline__ unsigned short f2bf(float f) {
    unsigned u = __builtin_bit_cast(unsigned, f);
    unsigned r = u + 0x7fffu + ((u >> 16) & 1u);   // RNE
    return (unsigned short)(r >> 16);
}
__device__ __forceinline__ float bf2f(unsigned short s) {
    return __builtin_bit_cast(float, ((unsigned)s) << 16);
}

__device__ __forceinline__ v8f wmma_bf16(v16bf a, v16bf b, v8f c) {
    return __builtin_amdgcn_wmma_f32_16x16x32_bf16(false, a, false, b, (short)0, c, false, false);
}

// ---- Convert a row-major fp32 weight [M x Kdim] into WMMA A-fragment bf16 layout.
__global__ void conv_wfrag(const float* __restrict__ W, unsigned short* __restrict__ frag,
                           int Kdim, int KT) {
    int blk = blockIdx.x;
    int kt  = blk % KT;
    int mt  = blk / KT;
    int lane = threadIdx.x;
    int m  = mt * 16 + (lane & 15);
    int kb = kt * 32 + (lane & 16);
    unsigned short* dst = frag + ((size_t)blk * 32 + lane) * 16;
#pragma unroll
    for (int e = 0; e < 16; ++e) {
        int k = kb + e;
        dst[e] = f2bf(W[(size_t)m * Kdim + k]);
    }
}

// ---- Convert x_t [D x B] slices into B-fragment layout for all t.
__global__ void conv_xfrag(const float* __restrict__ inputs, unsigned short* __restrict__ XF) {
    int blk = blockIdx.x;                 // (t*KT_D + kt)*NB + nb
    int nb = blk % NB;
    int kt = (blk / NB) % KT_D;
    int t  = blk / (NB * KT_D);
    int lane = threadIdx.x;
    const float* S = inputs + (size_t)t * D_DIM * B_DIM;
    int col = nb * 16 + (lane & 15);
    int kb  = kt * 32 + (lane & 16);
    unsigned short* dst = XF + (size_t)blk * FRAG + lane * 16;
#pragma unroll
    for (int e = 0; e < 16; ++e)
        dst[e] = f2bf(S[(size_t)(kb + e) * B_DIM + col]);
}

// ---- Convert h0 [H x B] into B-fragment layout HF[(kt*NB+nb)]
__global__ void conv_h0(const float* __restrict__ h0, unsigned short* __restrict__ HF) {
    int blk = blockIdx.x;                 // kt*NB + nb
    int nb = blk % NB;
    int kt = blk / NB;
    int lane = threadIdx.x;
    int col = nb * 16 + (lane & 15);
    int kb  = kt * 32 + (lane & 16);
    unsigned short* dst = HF + (size_t)blk * FRAG + lane * 16;
#pragma unroll
    for (int e = 0; e < 16; ++e)
        dst[e] = f2bf(h0[(size_t)(kb + e) * B_DIM + col]);
}

// ---- Export final h fragments back to plain [H x B] fp32
__global__ void export_h(const unsigned short* __restrict__ HF, float* __restrict__ outh) {
    int blk = blockIdx.x;                 // kt*NB + nb
    int nb = blk % NB;
    int kt = blk / NB;
    int lane = threadIdx.x;
    int col = nb * 16 + (lane & 15);
    int kb  = kt * 32 + (lane & 16);
    const unsigned short* src = HF + (size_t)blk * FRAG + lane * 16;
#pragma unroll
    for (int e = 0; e < 16; ++e)
        outh[(size_t)(kb + e) * B_DIM + col] = bf2f(src[e]);
}

// ---- One RNN timestep. Grid = NB(8) workgroups, each owns 16 batch columns.
__global__ __launch_bounds__(512) void rnn_step(
    const unsigned short* __restrict__ WhxF,
    const unsigned short* __restrict__ WhhF,
    const unsigned short* __restrict__ WhyF,
    const unsigned short* __restrict__ XF,
    unsigned short*       __restrict__ HF,
    const float* __restrict__ bh, const float* __restrict__ by,
    float* __restrict__ out, int t)
{
    __shared__ __align__(32) unsigned short hN[KT_H * FRAG]; // h_new in B-frag layout, 32KB
    __shared__ float wred[16][16];
    __shared__ float cmax[16];
    __shared__ float cinv[16];

    const int tid  = threadIdx.x;
    const int lane = tid & 31;
    const int wave = tid >> 5;            // 16 waves
    const int nb   = blockIdx.x;          // batch column block
    const int n    = lane & 15;

    // pull next step's x fragments toward this WGP while we grind the recurrence
    if (t + 1 < T_STEPS && tid < 64) {
        const unsigned short* p = XF +
            ((size_t)(t + 1) * KT_D * NB + (size_t)(tid >> 3) * NB + nb) * FRAG + (tid & 7) * 64;
        __builtin_prefetch(p, 0, 0);
    }

    // ---------- Phase 1: h_new = tanh(Whx@x_t + Whh@h + bh) ----------
    v8f acc[4] = {};                      // mt = wave*4 + i
    const unsigned short* xbase = XF + ((size_t)t * KT_D * NB + nb) * FRAG;

    { // Whx @ x_t  (K = 256), register double-buffered
        v16bf bf = *(const v16bf*)(xbase + (size_t)0 * NB * FRAG + lane * 16);
        v16bf af[4];
#pragma unroll
        for (int i = 0; i < 4; ++i)
            af[i] = *(const v16bf*)(WhxF + ((size_t)((wave * 4 + i) * KT_D) * 32 + lane) * 16);
        for (int kt = 0; kt < KT_D - 1; ++kt) {
            v16bf bn = *(const v16bf*)(xbase + (size_t)(kt + 1) * NB * FRAG + lane * 16);
            v16bf an[4];
#pragma unroll
            for (int i = 0; i < 4; ++i)
                an[i] = *(const v16bf*)(WhxF + ((size_t)((wave * 4 + i) * KT_D + kt + 1) * 32 + lane) * 16);
#pragma unroll
            for (int i = 0; i < 4; ++i)
                acc[i] = wmma_bf16(af[i], bf, acc[i]);
#pragma unroll
            for (int i = 0; i < 4; ++i) af[i] = an[i];
            bf = bn;
        }
#pragma unroll
        for (int i = 0; i < 4; ++i)
            acc[i] = wmma_bf16(af[i], bf, acc[i]);
    }

    { // Whh @ h  (K = 1024), register double-buffered
        v16bf bf = *(const v16bf*)(HF + ((size_t)(0 * NB + nb) * 32 + lane) * 16);
        v16bf af[4];
#pragma unroll
        for (int i = 0; i < 4; ++i)
            af[i] = *(const v16bf*)(WhhF + ((size_t)((wave * 4 + i) * KT_H) * 32 + lane) * 16);
        for (int kt = 0; kt < KT_H - 1; ++kt) {
            v16bf bn = *(const v16bf*)(HF + ((size_t)((kt + 1) * NB + nb) * 32 + lane) * 16);
            v16bf an[4];
#pragma unroll
            for (int i = 0; i < 4; ++i)
                an[i] = *(const v16bf*)(WhhF + ((size_t)((wave * 4 + i) * KT_H + kt + 1) * 32 + lane) * 16);
#pragma unroll
            for (int i = 0; i < 4; ++i)
                acc[i] = wmma_bf16(af[i], bf, acc[i]);
#pragma unroll
            for (int i = 0; i < 4; ++i) af[i] = an[i];
            bf = bn;
        }
#pragma unroll
        for (int i = 0; i < 4; ++i)
            acc[i] = wmma_bf16(af[i], bf, acc[i]);
    }

    // tanh + bias, scatter h_new into LDS in B-fragment layout
#pragma unroll
    for (int i = 0; i < 4; ++i) {
        int mt = wave * 4 + i;
#pragma unroll
        for (int r = 0; r < 8; ++r) {
            int g = mt * 16 + r + ((lane < 16) ? 0 : 8);     // global H row
            float v = tanhf(acc[i][r] + bh[g]);
            int kt2 = g >> 5;
            int rin = g & 31;
            int lb  = (rin < 16) ? n : (16 + n);
            int e   = rin & 15;
            hN[(kt2 * 32 + lb) * 16 + e] = f2bf(v);
        }
    }
    __syncthreads();

    // persist h_new fragments for next step (each WG owns its nb strip -> in-place safe)
    for (int j = tid; j < KT_H * FRAG / 8; j += 512) {       // 2048 uint4 total
        int kt = j >> 6;
        int w  = j & 63;
        ((uint4*)(HF + (size_t)(kt * NB + nb) * FRAG))[w] = ((const uint4*)hN)[j];
    }

    // ---------- Phase 2: y = Why @ h_new, register double-buffered ----------
    v8f acc2[4] = {};                     // ct = wave*4 + i
    {
        v16bf bf = *(const v16bf*)&hN[(0 * 32 + lane) * 16];
        v16bf af[4];
#pragma unroll
        for (int i = 0; i < 4; ++i)
            af[i] = *(const v16bf*)(WhyF + ((size_t)((wave * 4 + i) * KT_H) * 32 + lane) * 16);
        for (int kt = 0; kt < KT_H - 1; ++kt) {
            v16bf bn = *(const v16bf*)&hN[((kt + 1) * 32 + lane) * 16];
            v16bf an[4];
#pragma unroll
            for (int i = 0; i < 4; ++i)
                an[i] = *(const v16bf*)(WhyF + ((size_t)((wave * 4 + i) * KT_H + kt + 1) * 32 + lane) * 16);
#pragma unroll
            for (int i = 0; i < 4; ++i)
                acc2[i] = wmma_bf16(af[i], bf, acc2[i]);
#pragma unroll
            for (int i = 0; i < 4; ++i) af[i] = an[i];
            bf = bn;
        }
#pragma unroll
        for (int i = 0; i < 4; ++i)
            acc2[i] = wmma_bf16(af[i], bf, acc2[i]);
    }

    // y values stay in registers; each lane holds 32 values, all of column n
    float yv[4][8];
    float lmax = -3.402823466e38f;
#pragma unroll
    for (int i = 0; i < 4; ++i) {
#pragma unroll
        for (int r = 0; r < 8; ++r) {
            int c = (wave * 4 + i) * 16 + r + ((lane < 16) ? 0 : 8);
            float y = acc2[i][r] + by[c];
            yv[i][r] = y;
            lmax = fmaxf(lmax, y);
        }
    }
    // ---------- softmax over C (column-wise) ----------
    lmax = fmaxf(lmax, __shfl_xor(lmax, 16));   // merge lane halves (same column)
    if (lane < 16) wred[wave][lane] = lmax;
    __syncthreads();
    if (tid < 16) {
        float m = wred[0][tid];
#pragma unroll
        for (int w = 1; w < 16; ++w) m = fmaxf(m, wred[w][tid]);
        cmax[tid] = m;
    }
    __syncthreads();
    float cm = cmax[n];
    float lsum = 0.f;
#pragma unroll
    for (int i = 0; i < 4; ++i)
#pragma unroll
        for (int r = 0; r < 8; ++r) {
            float e = __expf(yv[i][r] - cm);
            yv[i][r] = e;
            lsum += e;
        }
    lsum += __shfl_xor(lsum, 16);
    if (lane < 16) wred[wave][lane] = lsum;
    __syncthreads();
    if (tid < 16) {
        float s = 0.f;
#pragma unroll
        for (int w = 0; w < 16; ++w) s += wred[w][tid];
        cinv[tid] = 1.f / s;
    }
    __syncthreads();
    float ci = cinv[n];

    // out[t][b][c]: lane writes 4 * 8-consecutive-float runs for its batch row
    size_t ob = ((size_t)t * B_DIM + nb * 16 + n) * C_DIM;
#pragma unroll
    for (int i = 0; i < 4; ++i) {
        int c0 = (wave * 4 + i) * 16 + ((lane < 16) ? 0 : 8);
#pragma unroll
        for (int r = 0; r < 8; ++r)
            out[ob + c0 + r] = yv[i][r] * ci;
    }
}

extern "C" void kernel_launch(void* const* d_in, const int* in_sizes, int n_in,
                              void* d_out, int out_size, void* d_ws, size_t ws_size,
                              hipStream_t stream) {
    (void)in_sizes; (void)n_in; (void)out_size; (void)ws_size;
    const float* inputs = (const float*)d_in[0];
    const float* h0     = (const float*)d_in[1];
    const float* Whx    = (const float*)d_in[2];
    const float* Whh    = (const float*)d_in[3];
    const float* Why    = (const float*)d_in[4];
    const float* bh     = (const float*)d_in[5];
    const float* by     = (const float*)d_in[6];
    float* out = (float*)d_out;

    unsigned short* ws   = (unsigned short*)d_ws;
    unsigned short* WhxF = ws + WS_WHX;
    unsigned short* WhhF = ws + WS_WHH;
    unsigned short* WhyF = ws + WS_WHY;
    unsigned short* XF   = ws + WS_XF;
    unsigned short* HF   = ws + WS_HF;

    // one-time (per call) operand swizzle into bf16 WMMA fragment layouts
    conv_wfrag<<<MT_H * KT_D, 32, 0, stream>>>(Whx, WhxF, D_DIM, KT_D);
    conv_wfrag<<<MT_H * KT_H, 32, 0, stream>>>(Whh, WhhF, H_DIM, KT_H);
    conv_wfrag<<<MT_H * KT_H, 32, 0, stream>>>(Why, WhyF, H_DIM, KT_H);
    conv_xfrag<<<T_STEPS * KT_D * NB, 32, 0, stream>>>(inputs, XF);
    conv_h0<<<KT_H * NB, 32, 0, stream>>>(h0, HF);

    // sequential recurrence: one fused kernel per timestep
    for (int t = 0; t < T_STEPS; ++t)
        rnn_step<<<NB, 512, 0, stream>>>(WhxF, WhhF, WhyF, XF, HF, bh, by, out, t);

    export_h<<<KT_H * NB, 32, 0, stream>>>(HF, out + (size_t)T_STEPS * B_DIM * C_DIM);
}
